// Genotype2PhenotypeTransformer_89404039233968
// MI455X (gfx1250) — compile-verified, matrix operand order checked by reading, not run on hardware
//
#include <hip/hip_runtime.h>
#include <cmath>

typedef __bf16 bf16;
typedef __attribute__((ext_vector_type(16))) __bf16 v16bf;
typedef __attribute__((ext_vector_type(8)))  float  v8f;

constexpr int Bb   = 16;
constexpr int NSYSs= 2048;
constexpr int Dm   = 512;
constexpr int Hh   = 4;
constexpr int DHh  = 128;
constexpr int Ll   = 6;
constexpr int NQq  = 512;
constexpr int NKk  = 512;
constexpr int FFf  = 2048;

// ---------------------------------------------------------------------------
// WMMA fragment load: 16-bit A/B fragment, 16x32 tile, k-contiguous storage.
// Lane L: row = row0 + (L&15); group g = L>>4.
// elements 0..7  = [k0 + g*8 + 0..7]
// elements 8..15 = [k0 + 16 + g*8 + 0..7]
// => two 16-byte global_load_b128 per lane.
// ---------------------------------------------------------------------------
union FragU { v16bf v; uint4 q[2]; };

__device__ inline v16bf load_frag(const bf16* __restrict__ base, int ld,
                                  int row0, int k0, int lane) {
  int r = row0 + (lane & 15);
  int g = lane >> 4;
  const bf16* p = base + (size_t)r * ld + k0 + g * 8;
  FragU u;
  u.q[0] = *(const uint4*)(p);
  u.q[1] = *(const uint4*)(p + 16);
  return u.v;
}

// ---------------------------------------------------------------------------
// Generic batched bf16 GEMM: C[M,N] = alpha * A[M,K] @ Bt[N,K]^T
// Batch dims: blockIdx.y (outer, e.g. batch) and blockIdx.z (e.g. head),
// with independent element strides for A, Bt, C.
// mode: 0 = f32 out, 1 = bf16 out, 2 = gelu->bf16 out,
//       3 = bf16 transposed out (Vt: addr = (m/tprows)*(N*tprows)+n*tprows+m%tprows)
// Wave tile 32x64 (2x4 of 16x16), block = 8 waves.
// ---------------------------------------------------------------------------
__global__ void __launch_bounds__(256) wmma_gemm_kernel(
    const bf16* __restrict__ A,  int lda, long long sAb, long long sAh,
    const bf16* __restrict__ Bt, int ldb, long long sBb, long long sBh,
    void* __restrict__ Cout,     int ldc, long long sCb, long long sCh,
    int M, int N, int K, float alpha, int mode, int tprows)
{
  int lane  = threadIdx.x & 31;
  int wave  = blockIdx.x * (blockDim.x >> 5) + (threadIdx.x >> 5);
  int tilesM = M >> 5;
  int tilesN = N >> 6;
  if (wave >= tilesM * tilesN) return;           // wave-uniform: EXEC stays all-1s
  int tm = wave % tilesM;
  int tn = wave / tilesM;

  A  += (size_t)blockIdx.y * sAb + (size_t)blockIdx.z * sAh;
  Bt += (size_t)blockIdx.y * sBb + (size_t)blockIdx.z * sBh;
  size_t cbase = (size_t)blockIdx.y * sCb + (size_t)blockIdx.z * sCh;

  int m0 = tm * 32, n0 = tn * 64;
  v8f acc[2][4] = {};

  for (int k0 = 0; k0 < K; k0 += 32) {
    v16bf a0 = load_frag(A,  lda, m0,      k0, lane);
    v16bf a1 = load_frag(A,  lda, m0 + 16, k0, lane);
    v16bf b0 = load_frag(Bt, ldb, n0,      k0, lane);
    v16bf b1 = load_frag(Bt, ldb, n0 + 16, k0, lane);
    v16bf b2 = load_frag(Bt, ldb, n0 + 32, k0, lane);
    v16bf b3 = load_frag(Bt, ldb, n0 + 48, k0, lane);
    acc[0][0] = __builtin_amdgcn_wmma_f32_16x16x32_bf16(false, a0, false, b0, (short)0, acc[0][0], false, false);
    acc[0][1] = __builtin_amdgcn_wmma_f32_16x16x32_bf16(false, a0, false, b1, (short)0, acc[0][1], false, false);
    acc[0][2] = __builtin_amdgcn_wmma_f32_16x16x32_bf16(false, a0, false, b2, (short)0, acc[0][2], false, false);
    acc[0][3] = __builtin_amdgcn_wmma_f32_16x16x32_bf16(false, a0, false, b3, (short)0, acc[0][3], false, false);
    acc[1][0] = __builtin_amdgcn_wmma_f32_16x16x32_bf16(false, a1, false, b0, (short)0, acc[1][0], false, false);
    acc[1][1] = __builtin_amdgcn_wmma_f32_16x16x32_bf16(false, a1, false, b1, (short)0, acc[1][1], false, false);
    acc[1][2] = __builtin_amdgcn_wmma_f32_16x16x32_bf16(false, a1, false, b2, (short)0, acc[1][2], false, false);
    acc[1][3] = __builtin_amdgcn_wmma_f32_16x16x32_bf16(false, a1, false, b3, (short)0, acc[1][3], false, false);
  }

  // C/D 16x16 f32 layout: VGPR v, lanes 0-15 -> M=v,N=lane; lanes 16-31 -> M=8+v
  int rr = (lane >> 4) * 8;
  int cc = lane & 15;
  #pragma unroll
  for (int i = 0; i < 2; i++)
    #pragma unroll
    for (int j = 0; j < 4; j++)
      #pragma unroll
      for (int v = 0; v < 8; v++) {
        int gm = m0 + i * 16 + rr + v;
        int gn = n0 + j * 16 + cc;
        float val = acc[i][j][v] * alpha;
        if (mode == 0) {
          ((float*)Cout)[cbase + (size_t)gm * ldc + gn] = val;
        } else if (mode == 1) {
          ((bf16*)Cout)[cbase + (size_t)gm * ldc + gn] = (bf16)val;
        } else if (mode == 2) {
          float u = 0.7978845608028654f * (val + 0.044715f * val * val * val);
          float g = 0.5f * val * (1.0f + tanhf(u));
          ((bf16*)Cout)[cbase + (size_t)gm * ldc + gn] = (bf16)g;
        } else {
          size_t addr = (size_t)(gm / tprows) * ((size_t)N * tprows)
                      + (size_t)gn * tprows + (size_t)(gm % tprows);
          ((bf16*)Cout)[addr] = (bf16)val;
        }
      }
}

// ---------------------------------------------------------------------------
// Block reductions (block = 256 threads = 8 wave32s)
// ---------------------------------------------------------------------------
__device__ inline void block_red2(float& a, float& b, float* sa, float* sb) {
  #pragma unroll
  for (int o = 16; o > 0; o >>= 1) { a += __shfl_xor(a, o); b += __shfl_xor(b, o); }
  int w = threadIdx.x >> 5;
  if ((threadIdx.x & 31) == 0) { sa[w] = a; sb[w] = b; }
  __syncthreads();
  float ta = 0.f, tb = 0.f;
  #pragma unroll
  for (int i = 0; i < 8; i++) { ta += sa[i]; tb += sb[i]; }
  a = ta; b = tb;
  __syncthreads();
}

__device__ inline float block_redmax(float a, float* sh) {
  #pragma unroll
  for (int o = 16; o > 0; o >>= 1) a = fmaxf(a, __shfl_xor(a, o));
  if ((threadIdx.x & 31) == 0) sh[threadIdx.x >> 5] = a;
  __syncthreads();
  float r = sh[0];
  #pragma unroll
  for (int i = 1; i < 8; i++) r = fmaxf(r, sh[i]);
  __syncthreads();
  return r;
}

__device__ inline float block_redsum(float a, float* sh) {
  #pragma unroll
  for (int o = 16; o > 0; o >>= 1) a += __shfl_xor(a, o);
  if ((threadIdx.x & 31) == 0) sh[threadIdx.x >> 5] = a;
  __syncthreads();
  float r = 0.f;
  #pragma unroll
  for (int i = 0; i < 8; i++) r += sh[i];
  __syncthreads();
  return r;
}

// ---------------------------------------------------------------------------
// Gather + double layernorm: out_bf16[b,r,:] = LN(LN(table[idx[r]]) + update[b,idx[r]])
// eps = 1e-5 both. grid = (rows, B), block = 256 (each thread 2 of 512 elems).
// ---------------------------------------------------------------------------
__global__ void __launch_bounds__(256) prep_act_kernel(
    const float* __restrict__ table, const float* __restrict__ update,
    const int* __restrict__ idx, const float* __restrict__ sc,
    const float* __restrict__ bi, bf16* __restrict__ outp)
{
  __shared__ float sa[8], sb[8];
  int r = blockIdx.x, b = blockIdx.y, t = threadIdx.x;
  int gi = idx[r];
  const float* trow = table + (size_t)gi * Dm;
  const float* urow = update + ((size_t)b * NSYSs + (size_t)gi) * Dm;
  float x0 = trow[t], x1 = trow[t + 256];
  float s = x0 + x1, s2 = x0 * x0 + x1 * x1;
  block_red2(s, s2, sa, sb);
  float mean = s / Dm, var = s2 / Dm - mean * mean;
  float inv = rsqrtf(var + 1e-5f);
  float l0 = (x0 - mean) * inv * sc[t]       + bi[t];
  float l1 = (x1 - mean) * inv * sc[t + 256] + bi[t + 256];
  float y0 = l0 + urow[t], y1 = l1 + urow[t + 256];
  s = y0 + y1; s2 = y0 * y0 + y1 * y1;
  block_red2(s, s2, sa, sb);
  mean = s / Dm; var = s2 / Dm - mean * mean;
  inv = rsqrtf(var + 1e-5f);
  bf16* orow = outp + ((size_t)b * gridDim.x + r) * Dm;
  orow[t]       = (bf16)((y0 - mean) * inv * sc[t]       + bi[t]);
  orow[t + 256] = (bf16)((y1 - mean) * inv * sc[t + 256] + bi[t + 256]);
}

// ---------------------------------------------------------------------------
// LayerNorm: x = a[row] (+ b2[row] if non-null); writes f32 (if fout) and bf16
// (if bfout). grid = rows, block = 256.
// ---------------------------------------------------------------------------
__global__ void __launch_bounds__(256) ln_kernel(
    const float* __restrict__ a, const float* __restrict__ b2,
    const float* __restrict__ sc, const float* __restrict__ bi, float eps,
    float* __restrict__ fout, bf16* __restrict__ bfout)
{
  __shared__ float sa[8], sb[8];
  size_t row = blockIdx.x; int t = threadIdx.x;
  const float* ar = a + row * Dm;
  float x0 = ar[t], x1 = ar[t + 256];
  if (b2) { const float* br = b2 + row * Dm; x0 += br[t]; x1 += br[t + 256]; }
  float s = x0 + x1, s2 = x0 * x0 + x1 * x1;
  block_red2(s, s2, sa, sb);
  float mean = s / Dm, var = s2 / Dm - mean * mean;
  float inv = rsqrtf(var + eps);
  float o0 = (x0 - mean) * inv * sc[t]       + bi[t];
  float o1 = (x1 - mean) * inv * sc[t + 256] + bi[t + 256];
  if (fout)  { fout[row * Dm + t] = o0;       fout[row * Dm + t + 256] = o1; }
  if (bfout) { bfout[row * Dm + t] = (bf16)o0; bfout[row * Dm + t + 256] = (bf16)o1; }
}

// ---------------------------------------------------------------------------
// Masked softmax, in-place f32 -> bf16 (bf16 row packed in low half of the
// f32 row's bytes). grid = (NQ, H, B).
// ---------------------------------------------------------------------------
__global__ void __launch_bounds__(256) softmax_kernel(
    float* __restrict__ scores, const int* __restrict__ mask)
{
  __shared__ float sh[8];
  int q = blockIdx.x, h = blockIdx.y, b = blockIdx.z, t = threadIdx.x;
  size_t row = (((size_t)b * Hh + h) * NQq + q);
  float* srow = scores + row * NKk;
  const int* mrow = mask + (size_t)q * NKk;
  float v0 = (mrow[t] > 0)       ? srow[t]       : -1e9f;
  float v1 = (mrow[t + 256] > 0) ? srow[t + 256] : -1e9f;
  float mx = block_redmax(fmaxf(v0, v1), sh);
  float e0 = __expf(v0 - mx), e1 = __expf(v1 - mx);
  float inv = 1.0f / block_redsum(e0 + e1, sh);
  bf16* arow = (bf16*)srow;        // reads above complete before barrier
  arow[t]       = (bf16)(e0 * inv);
  arow[t + 256] = (bf16)(e1 * inv);
}

// ---------------------------------------------------------------------------
// Weight convert+transpose: W[K,N] f32 -> Wt[N,K] bf16
// ---------------------------------------------------------------------------
__global__ void wconv_kernel(const float* __restrict__ W, bf16* __restrict__ Wt,
                             int Kd, int Nd) {
  size_t i = (size_t)blockIdx.x * blockDim.x + threadIdx.x;
  if (i >= (size_t)Kd * Nd) return;
  int n = (int)(i / Kd);
  int k = (int)(i % Kd);
  Wt[i] = (bf16)W[(size_t)k * Nd + n];
}

__global__ void zero_kernel(float* __restrict__ p, size_t n) {
  size_t i = (size_t)blockIdx.x * blockDim.x + threadIdx.x;
  if (i < n) p[i] = 0.f;
}

__global__ void scale2_kernel(float* __restrict__ p, size_t n) {
  size_t i = (size_t)blockIdx.x * blockDim.x + threadIdx.x;
  if (i < n) p[i] *= 2.0f;
}

// update[b, idx[r], :] += eff[b, r, :]  (indices may repeat -> f32 HW atomics)
__global__ void __launch_bounds__(256) scatter_kernel(
    float* __restrict__ update, const float* __restrict__ eff,
    const int* __restrict__ idx)
{
  int r = blockIdx.x, b = blockIdx.y, t = threadIdx.x;
  int gi = idx[r];
  float* urow = update + ((size_t)b * NSYSs + (size_t)gi) * Dm;
  const float* erow = eff + ((size_t)b * NQq + r) * Dm;
  unsafeAtomicAdd(&urow[t], erow[t]);
  unsafeAtomicAdd(&urow[t + 256], erow[t + 256]);
}

__global__ void addout_kernel(const float* __restrict__ a,
                              const float* __restrict__ u,
                              float* __restrict__ o, size_t n) {
  size_t i = (size_t)blockIdx.x * blockDim.x + threadIdx.x;
  if (i < n) o[i] = a[i] + u[i];
}

// ---------------------------------------------------------------------------
// Host side
// ---------------------------------------------------------------------------
static inline void launch_gemm(hipStream_t st,
    const bf16* A, int lda, long long sAb, long long sAh,
    const bf16* Bt, int ldb, long long sBb, long long sBh,
    void* C, int ldc, long long sCb, long long sCh,
    int M, int N, int K, float alpha, int mode, int tprows, int batB, int batH)
{
  int tiles = (M / 32) * (N / 64);
  dim3 g((tiles + 7) / 8, batB, batH);
  wmma_gemm_kernel<<<g, 256, 0, st>>>(A, lda, sAb, sAh, Bt, ldb, sBb, sBh,
                                      C, ldc, sCb, sCh, M, N, K, alpha, mode, tprows);
}

extern "C" void kernel_launch(void* const* d_in, const int* in_sizes, int n_in,
                              void* d_out, int out_size, void* d_ws, size_t ws_size,
                              hipStream_t stream) {
  (void)in_sizes; (void)n_in; (void)out_size; (void)ws_size;
  const float* sysemb = (const float*)d_in[0];
  const float* table  = (const float*)d_in[1];
  const float* sscale = (const float*)d_in[2];
  const float* sbias  = (const float*)d_in[3];
  const float* Wq     = (const float*)d_in[4];
  const float* Wk     = (const float*)d_in[5];
  const float* Wv     = (const float*)d_in[6];
  const float* Wo     = (const float*)d_in[7];
  const float* W1     = (const float*)d_in[8];
  const float* W2     = (const float*)d_in[9];
  const float* nis    = (const float*)d_in[10];
  const float* nib    = (const float*)d_in[11];
  const float* nos    = (const float*)d_in[12];
  const float* nob    = (const float*)d_in[13];
  const int* qidx     = (const int*)d_in[14];
  const int* kidx     = (const int*)d_in[15];
  const int* masks    = (const int*)d_in[16];
  float* out = (float*)d_out;
  char* ws = (char*)d_ws;

  size_t off = 0;
  auto alloc = [&](size_t bytes) { size_t r = off; off += (bytes + 255) & ~(size_t)255; return r; };
  bf16* Wqt = (bf16*)(ws + alloc((size_t)Dm * Dm * 2));
  bf16* Wkt = (bf16*)(ws + alloc((size_t)Dm * Dm * 2));
  bf16* Wvt = (bf16*)(ws + alloc((size_t)Dm * Dm * 2));
  bf16* Wot = (bf16*)(ws + alloc((size_t)Dm * Dm * 2));
  bf16* W1t = (bf16*)(ws + alloc((size_t)Dm * FFf * 2));   // [FF, D]
  bf16* W2t = (bf16*)(ws + alloc((size_t)FFf * Dm * 2));   // [D, FF]
  float* update = (float*)(ws + alloc((size_t)Bb * NSYSs * Dm * 4));
  bf16* qbuf = (bf16*)(ws + alloc((size_t)Bb * NQq * Dm * 2));
  bf16* kbuf = (bf16*)(ws + alloc((size_t)Bb * NKk * Dm * 2));
  bf16* Qb   = (bf16*)(ws + alloc((size_t)Bb * NQq * Dm * 2));
  bf16* Kb   = (bf16*)(ws + alloc((size_t)Bb * NKk * Dm * 2));
  bf16* Vt   = (bf16*)(ws + alloc((size_t)Bb * Dm * NKk * 2)); // [B, H*DH, NK]
  float* big = (float*)(ws + alloc((size_t)Bb * Hh * NQq * NKk * 4)); // scores/attn; later h1
  bf16* ctxb = (bf16*)(ws + alloc((size_t)Bb * NQq * Dm * 2));
  float* t0  = (float*)(ws + alloc((size_t)Bb * NQq * Dm * 4));
  float* yf  = (float*)(ws + alloc((size_t)Bb * NQq * Dm * 4));
  bf16* ybf  = (bf16*)(ws + alloc((size_t)Bb * NQq * Dm * 2));
  float* eff = (float*)(ws + alloc((size_t)Bb * NQq * Dm * 4));

  const int M  = Bb * NQq;                 // 8192
  const size_t NU = (size_t)Bb * NSYSs * Dm;

  // One-time per call: weight bf16 transposes + zero update
  {
    size_t n = (size_t)Dm * Dm;
    dim3 g((unsigned)((n + 255) / 256));
    wconv_kernel<<<g, 256, 0, stream>>>(Wq, Wqt, Dm, Dm);
    wconv_kernel<<<g, 256, 0, stream>>>(Wk, Wkt, Dm, Dm);
    wconv_kernel<<<g, 256, 0, stream>>>(Wv, Wvt, Dm, Dm);
    wconv_kernel<<<g, 256, 0, stream>>>(Wo, Wot, Dm, Dm);
    size_t nf = (size_t)Dm * FFf;
    dim3 gf((unsigned)((nf + 255) / 256));
    wconv_kernel<<<gf, 256, 0, stream>>>(W1, W1t, Dm, FFf);
    wconv_kernel<<<gf, 256, 0, stream>>>(W2, W2t, FFf, Dm);
    zero_kernel<<<(unsigned)((NU + 255) / 256), 256, 0, stream>>>(update, NU);
  }

  const float inv_sqrt_dh = 0.08838834764831845f; // 1/sqrt(128)

  for (int l = 0; l < Ll; l++) {
    const int* qi = qidx + (size_t)l * NQq;
    const int* ki = kidx + (size_t)l * NKk;
    const int* mk = masks + (size_t)l * NQq * NKk;

    // q = LN(LN(table[qi]) + update[:,qi]); k likewise
    prep_act_kernel<<<dim3(NQq, Bb), 256, 0, stream>>>(table, update, qi, sscale, sbias, qbuf);
    prep_act_kernel<<<dim3(NKk, Bb), 256, 0, stream>>>(table, update, ki, sscale, sbias, kbuf);

    // Q = q@Wq, K = k@Wk  (bf16 out), V = k@Wv stored transposed [B,H*DH,NK]
    launch_gemm(stream, qbuf, Dm, 0, 0, Wqt, Dm, 0, 0, Qb, Dm, 0, 0, M, Dm, Dm, 1.f, 1, 0, 1, 1);
    launch_gemm(stream, kbuf, Dm, 0, 0, Wkt, Dm, 0, 0, Kb, Dm, 0, 0, M, Dm, Dm, 1.f, 1, 0, 1, 1);
    launch_gemm(stream, kbuf, Dm, 0, 0, Wvt, Dm, 0, 0, Vt, 0, 0, 0, M, Dm, Dm, 1.f, 3, NKk, 1, 1);

    // scores[b,h,q,k] = (Q . K) / sqrt(DH)   batched over (b,h)
    launch_gemm(stream,
        Qb, Dm, (long long)NQq * Dm, DHh,
        Kb, Dm, (long long)NKk * Dm, DHh,
        big, NKk, (long long)Hh * NQq * NKk, (long long)NQq * NKk,
        NQq, NKk, DHh, inv_sqrt_dh, 0, 0, Bb, Hh);

    // masked softmax (in-place f32 -> bf16)
    softmax_kernel<<<dim3(NQq, Hh, Bb), 256, 0, stream>>>(big, mk);

    // ctx[b,q,h*DH+dh] = attn @ V     (attn bf16 lives inside `big`, lda = 2*NK)
    launch_gemm(stream,
        (const bf16*)big, 2 * NKk, (long long)Hh * NQq * 2 * NKk, (long long)NQq * 2 * NKk,
        Vt, NKk, (long long)Dm * NKk, (long long)DHh * NKk,
        ctxb, Dm, (long long)NQq * Dm, DHh,
        NQq, DHh, NKk, 1.f, 1, 0, Bb, Hh);

    // t0 = ctx @ Wo  (f32) ; y = LN(t0, ni, eps=0.1) -> yf (f32) + ybf (bf16)
    launch_gemm(stream, ctxb, Dm, 0, 0, Wot, Dm, 0, 0, t0, Dm, 0, 0, M, Dm, Dm, 1.f, 0, 0, 1, 1);
    ln_kernel<<<M, 256, 0, stream>>>(t0, nullptr, nis, nib, 0.1f, yf, ybf);

    // h1 = gelu(y @ W1) bf16 (reuses `big`); t0 = h1 @ W2 (f32)
    launch_gemm(stream, ybf, Dm, 0, 0, W1t, Dm, 0, 0, big, FFf, 0, 0, M, FFf, Dm, 1.f, 2, 0, 1, 1);
    launch_gemm(stream, (const bf16*)big, FFf, 0, 0, W2t, FFf, 0, 0, t0, Dm, 0, 0, M, Dm, FFf, 1.f, 0, 0, 1, 1);

    // eff = LN(y + ffn, no, eps=0.1)
    ln_kernel<<<M, 256, 0, stream>>>(yf, t0, nos, nob, 0.1f, eff, nullptr);

    // update = 2*update + scatter_add(eff at qi)
    scale2_kernel<<<(unsigned)((NU + 255) / 256), 256, 0, stream>>>(update, NU);
    scatter_kernel<<<dim3(NQq, Bb), 256, 0, stream>>>(update, eff, qi);
  }

  addout_kernel<<<(unsigned)((NU + 255) / 256), 256, 0, stream>>>(sysemb, update, out, NU);
}
